// AttentionBlock_67250597920915
// MI455X (gfx1250) — compile-verified, matrix operand order checked by reading, not run on hardware
//
#include <hip/hip_runtime.h>
#include <stdint.h>

// ---------------- problem constants ----------------
#define D_EMB   1024
#define T_SEQ   4096
#define B_SZ    8
#define NCHUNK  32
#define CHUNK   128              // T_SEQ / NCHUNK
#define M_TOT   (B_SZ * T_SEQ)   // 32768 rows

// ---------------- WMMA types ----------------
typedef __attribute__((ext_vector_type(16))) __bf16 v16bf;
typedef __attribute__((ext_vector_type(8)))  float  v8f;
typedef int v4i_vs __attribute__((vector_size(16)));   // matches builtin param type

typedef unsigned short bf16_t;   // raw bf16 storage

#if __has_builtin(__builtin_amdgcn_global_load_async_to_lds_b128)
#define HAVE_ASYNC_LDS 1
#else
#define HAVE_ASYNC_LDS 0
#endif

__device__ __forceinline__ void wait_async_zero() {
#if __has_builtin(__builtin_amdgcn_s_wait_asynccnt)
  __builtin_amdgcn_s_wait_asynccnt(0);
#else
  asm volatile("s_wait_asynccnt 0x0" ::: "memory");
#endif
}

__device__ __forceinline__ bf16_t f32_to_bf16(float f) {
  union { float f; unsigned int u; } c; c.f = f;
  unsigned int lsb = (c.u >> 16) & 1u;
  c.u += 0x7fffu + lsb;                 // round-to-nearest-even
  return (bf16_t)(c.u >> 16);
}

// =====================================================================
// fp32 -> bf16 cast (weights)
// =====================================================================
__global__ __launch_bounds__(256)
void cast_f32_bf16(const float* __restrict__ s, bf16_t* __restrict__ d, int n) {
  int i = blockIdx.x * 256 + threadIdx.x;
  if (i < n) d[i] = f32_to_bf16(s[i]);
}

// =====================================================================
// token shift: xk/xv/xr = mix*x_t + (1-mix)*x_{t-1}, bf16 out
// =====================================================================
__global__ __launch_bounds__(256)
void token_shift(const float* __restrict__ x,
                 const float* __restrict__ mk, const float* __restrict__ mv,
                 const float* __restrict__ mr,
                 bf16_t* __restrict__ xk, bf16_t* __restrict__ xv,
                 bf16_t* __restrict__ xr) {
  size_t gid = (size_t)blockIdx.x * 256 + threadIdx.x;
  int d = (int)(gid & (D_EMB - 1));
  int t = (int)((gid >> 10) & (T_SEQ - 1));
  float xc = x[gid];
  float xp = (t == 0) ? 0.0f : x[gid - D_EMB];
  float a;
  a = mk[d]; xk[gid] = f32_to_bf16(a * xc + (1.0f - a) * xp);
  a = mv[d]; xv[gid] = f32_to_bf16(a * xc + (1.0f - a) * xp);
  a = mr[d]; xr[gid] = f32_to_bf16(a * xc + (1.0f - a) * xp);
}

// =====================================================================
// bf16 WMMA GEMM: C[M,N](f32) = A[M,K](bf16) @ W[N,K](bf16)^T
// workgroup tile 128x128, 8 waves, each wave 64x32 (4x2 wmma tiles).
// Global->LDS via CDNA5 async-to-LDS (no VGPR staging) when available.
// =====================================================================
#define TILE_K          32
#define LDS_ROW_BYTES   80                    // 64B data + 16B pad (bank safe)
#define TILE_BYTES      (128 * LDS_ROW_BYTES) // 10240 per A or B tile
#define BUF_BYTES       (2 * TILE_BYTES)      // A + B

union FragBF { v16bf v; uint4 q[2]; };

__global__ __launch_bounds__(256, 1)
void gemm_bf16_wmma(const bf16_t* __restrict__ A, const bf16_t* __restrict__ W,
                    float* __restrict__ C, int M, int N, int K) {
  __shared__ __align__(16) unsigned char smem[2 * BUF_BYTES];  // double buffer

  const int tid  = threadIdx.x;
  const int lane = tid & 31;
  const int w    = tid >> 5;
  const int wm   = w >> 2;          // 0..1  (64 rows each)
  const int wn   = w & 3;           // 0..3  (32 cols each)
  const int rowBase = blockIdx.y * 128;
  const int colBase = blockIdx.x * 128;

  v8f acc[4][2];
#pragma unroll
  for (int i = 0; i < 4; ++i)
#pragma unroll
    for (int j = 0; j < 2; ++j)
#pragma unroll
      for (int r = 0; r < 8; ++r) acc[i][j][r] = 0.0f;

  // each thread moves 2 A chunks + 2 B chunks of 16B per K-step
  const int c0 = tid, c1 = tid + 256;
  const int rA0 = c0 >> 2, oA0 = (c0 & 3) << 4;
  const int rA1 = c1 >> 2, oA1 = (c1 & 3) << 4;
  const char* Abase = (const char*)A;
  const char* Wbase = (const char*)W;
  // per-thread global byte offsets (row part, constant over K loop)
  const size_t gA0 = ((size_t)(rowBase + rA0) * K) * 2 + oA0;
  const size_t gA1 = ((size_t)(rowBase + rA1) * K) * 2 + oA1;
  const size_t gB0 = ((size_t)(colBase + rA0) * K) * 2 + oA0;
  const size_t gB1 = ((size_t)(colBase + rA1) * K) * 2 + oA1;
  // per-thread LDS byte offsets within a tile
  const int lA0 = rA0 * LDS_ROW_BYTES + oA0;
  const int lA1 = rA1 * LDS_ROW_BYTES + oA1;

#if HAVE_ASYNC_LDS
  auto issueTile = [&](int buf, int kk) {
    const size_t kb = (size_t)kk * TILE_K * 2;
    unsigned char* As = smem + buf * BUF_BYTES;
    unsigned char* Bs = As + TILE_BYTES;
    __builtin_amdgcn_global_load_async_to_lds_b128(
        (v4i_vs*)(Abase + gA0 + kb), (v4i_vs*)(As + lA0), 0, 0);
    __builtin_amdgcn_global_load_async_to_lds_b128(
        (v4i_vs*)(Abase + gA1 + kb), (v4i_vs*)(As + lA1), 0, 0);
    __builtin_amdgcn_global_load_async_to_lds_b128(
        (v4i_vs*)(Wbase + gB0 + kb), (v4i_vs*)(Bs + lA0), 0, 0);
    __builtin_amdgcn_global_load_async_to_lds_b128(
        (v4i_vs*)(Wbase + gB1 + kb), (v4i_vs*)(Bs + lA1), 0, 0);
  };
#else
  uint4 ra0, ra1, rb0, rb1;   // scalar staging (no arrays -> no spill pressure)
  auto loadG = [&](int kk) {
    const size_t kb = (size_t)kk * TILE_K * 2;
    ra0 = *(const uint4*)(Abase + gA0 + kb);
    ra1 = *(const uint4*)(Abase + gA1 + kb);
    rb0 = *(const uint4*)(Wbase + gB0 + kb);
    rb1 = *(const uint4*)(Wbase + gB1 + kb);
  };
  auto storeL = [&](int buf) {
    unsigned char* As = smem + buf * BUF_BYTES;
    unsigned char* Bs = As + TILE_BYTES;
    *(uint4*)(As + lA0) = ra0;
    *(uint4*)(As + lA1) = ra1;
    *(uint4*)(Bs + lA0) = rb0;
    *(uint4*)(Bs + lA1) = rb1;
  };
#endif

  const int lrow = lane & 15;          // fragment row within 16-tile
  const int loff = (lane >> 4) << 4;   // 0 or 16 bytes (K split per ISA layout)

#if HAVE_ASYNC_LDS
  issueTile(0, 0);
  wait_async_zero();
  __syncthreads();
#else
  loadG(0);
  storeL(0);
  __syncthreads();
#endif

  const int KSTEPS = K / TILE_K;
  for (int kk = 0; kk < KSTEPS; ++kk) {
    const int cur = kk & 1;
    const bool hasNext = (kk + 1) < KSTEPS;

#if HAVE_ASYNC_LDS
    if (hasNext) issueTile(1 - cur, kk + 1);   // async fetch next tile into LDS
#else
    if (hasNext) loadG(kk + 1);
#endif

    const unsigned char* As = smem + cur * BUF_BYTES;
    const unsigned char* Bs = As + TILE_BYTES;

    FragBF af[4], bfr[2];
#pragma unroll
    for (int i = 0; i < 4; ++i) {
      const unsigned char* pa = As + (wm * 64 + i * 16 + lrow) * LDS_ROW_BYTES + loff;
      af[i].q[0] = *(const uint4*)pa;
      af[i].q[1] = *(const uint4*)(pa + 32);
    }
#pragma unroll
    for (int j = 0; j < 2; ++j) {
      const unsigned char* pb = Bs + (wn * 32 + j * 16 + lrow) * LDS_ROW_BYTES + loff;
      bfr[j].q[0] = *(const uint4*)pb;
      bfr[j].q[1] = *(const uint4*)(pb + 32);
    }

#pragma unroll
    for (int i = 0; i < 4; ++i)
#pragma unroll
      for (int j = 0; j < 2; ++j)
        acc[i][j] = __builtin_amdgcn_wmma_f32_16x16x32_bf16(
            false, af[i].v, false, bfr[j].v, (short)0, acc[i][j], false, false);

#if HAVE_ASYNC_LDS
    if (hasNext) wait_async_zero();            // my 4 async chunks landed
#else
    if (hasNext) storeL(1 - cur);
#endif
    __syncthreads();                           // everyone's chunks landed
  }

  // epilogue: D layout -> VGPR r holds M = r + 8*(lane>=16), N = lane&15
#pragma unroll
  for (int i = 0; i < 4; ++i)
#pragma unroll
    for (int j = 0; j < 2; ++j) {
      int m0 = rowBase + wm * 64 + i * 16 + ((lane >> 4) << 3);
      int n0 = colBase + wn * 32 + j * 16 + (lane & 15);
#pragma unroll
      for (int r = 0; r < 8; ++r)
        C[(size_t)(m0 + r) * N + n0] = acc[i][j][r];
    }
}

// =====================================================================
// WKV recurrence, chunked parallel scan over T
// =====================================================================
__global__ __launch_bounds__(256)
void wkv_pass1(const float* __restrict__ Kf, const float* __restrict__ Vf,
               const float* __restrict__ log_decay,
               float* __restrict__ Sc, float* __restrict__ Sn) {
  size_t gid = (size_t)blockIdx.x * 256 + threadIdx.x;
  int d = (int)(gid & (D_EMB - 1));
  int c = (int)((gid >> 10) & (NCHUNK - 1));
  int b = (int)(gid >> 15);
  float dec = __expf(-__expf(log_decay[d]));
  size_t base = ((size_t)b * T_SEQ + (size_t)c * CHUNK) * D_EMB + d;
  float cont = 0.0f, norm = 0.0f;
  for (int i = 0; i < CHUNK; ++i) {
    float k = Kf[base], v = Vf[base];
    float imp = __expf(k);
    cont = cont * dec + imp * v;
    norm = norm * dec + imp;
    base += D_EMB;
  }
  size_t s = ((size_t)b * NCHUNK + c) * D_EMB + d;
  Sc[s] = cont; Sn[s] = norm;
}

__global__ __launch_bounds__(256)
void wkv_combine(const float* __restrict__ Sc, const float* __restrict__ Sn,
                 const float* __restrict__ log_decay,
                 float* __restrict__ Ic, float* __restrict__ In) {
  int gid = blockIdx.x * 256 + threadIdx.x;
  int d = gid & (D_EMB - 1);
  int b = gid >> 10;
  float ed = __expf(log_decay[d]);
  float decL = __expf(-ed * (float)CHUNK);    // decay^CHUNK
  float cont = 0.0f, norm = 0.0f;
  for (int c = 0; c < NCHUNK; ++c) {
    size_t s = ((size_t)b * NCHUNK + c) * D_EMB + d;
    Ic[s] = cont; In[s] = norm;
    cont = cont * decL + Sc[s];
    norm = norm * decL + Sn[s];
  }
}

__global__ __launch_bounds__(256)
void wkv_pass2(const float* __restrict__ Kf, const float* __restrict__ Vf,
               const float* __restrict__ Rf,
               const float* __restrict__ log_decay, const float* __restrict__ log_gain,
               const float* __restrict__ Ic, const float* __restrict__ In,
               bf16_t* __restrict__ rwkv) {
  size_t gid = (size_t)blockIdx.x * 256 + threadIdx.x;
  int d = (int)(gid & (D_EMB - 1));
  int c = (int)((gid >> 10) & (NCHUNK - 1));
  int b = (int)(gid >> 15);
  float dec = __expf(-__expf(log_decay[d]));
  float g   = __expf(log_gain[d]) - 1.0f;
  size_t s = ((size_t)b * NCHUNK + c) * D_EMB + d;
  float cont = Ic[s], norm = In[s];
  size_t base = ((size_t)b * T_SEQ + (size_t)c * CHUNK) * D_EMB + d;
  for (int i = 0; i < CHUNK; ++i) {
    float k = Kf[base], v = Vf[base], r = Rf[base];
    float imp = __expf(k);
    cont = cont * dec + imp * v;
    norm = norm * dec + imp;
    float num = cont + g * imp * v;
    float den = norm + g * imp;
    float wkv = num / den;
    float sig = 1.0f / (1.0f + __expf(-r));
    rwkv[base] = f32_to_bf16(sig * wkv);
    base += D_EMB;
  }
}

// =====================================================================
// launch
// =====================================================================
extern "C" void kernel_launch(void* const* d_in, const int* in_sizes, int n_in,
                              void* d_out, int out_size, void* d_ws, size_t ws_size,
                              hipStream_t stream) {
  const float* x   = (const float*)d_in[0];
  const float* Wk  = (const float*)d_in[1];
  const float* Wv  = (const float*)d_in[2];
  const float* Wr  = (const float*)d_in[3];
  const float* Wo  = (const float*)d_in[4];
  const float* mk  = (const float*)d_in[5];
  const float* mv  = (const float*)d_in[6];
  const float* mr  = (const float*)d_in[7];
  const float* lg  = (const float*)d_in[8];
  const float* ld  = (const float*)d_in[9];
  float* out = (float*)d_out;

  // workspace carving
  char* p = (char*)d_ws;
  auto alloc = [&](size_t bytes) -> char* {
    char* r = p; p += (bytes + 255) & ~(size_t)255; return r;
  };
  const size_t WB = (size_t)D_EMB * D_EMB * sizeof(bf16_t);       // 2 MB
  const size_t XB = (size_t)M_TOT * D_EMB * sizeof(bf16_t);       // 64 MB
  const size_t FB = (size_t)M_TOT * D_EMB * sizeof(float);        // 128 MB
  const size_t SB = (size_t)B_SZ * NCHUNK * D_EMB * sizeof(float);// 1 MB

  bf16_t* wkb = (bf16_t*)alloc(WB);
  bf16_t* wvb = (bf16_t*)alloc(WB);
  bf16_t* wrb = (bf16_t*)alloc(WB);
  bf16_t* wob = (bf16_t*)alloc(WB);
  bf16_t* xk  = (bf16_t*)alloc(XB);
  bf16_t* xv  = (bf16_t*)alloc(XB);
  bf16_t* xr  = (bf16_t*)alloc(XB);
  float*  Kf  = (float*)alloc(FB);
  float*  Vf  = (float*)alloc(FB);
  float*  Rf  = (float*)alloc(FB);
  float*  Sc  = (float*)alloc(SB);
  float*  Sn  = (float*)alloc(SB);
  float*  Ic  = (float*)alloc(SB);
  float*  In  = (float*)alloc(SB);
  bf16_t* rwkv = xk;   // xk is dead after the K GEMM; reuse for rwkv

  const int WN = D_EMB * D_EMB;
  cast_f32_bf16<<<WN / 256, 256, 0, stream>>>(Wk, wkb, WN);
  cast_f32_bf16<<<WN / 256, 256, 0, stream>>>(Wv, wvb, WN);
  cast_f32_bf16<<<WN / 256, 256, 0, stream>>>(Wr, wrb, WN);
  cast_f32_bf16<<<WN / 256, 256, 0, stream>>>(Wo, wob, WN);

  const size_t TOT = (size_t)M_TOT * D_EMB;
  token_shift<<<(unsigned)(TOT / 256), 256, 0, stream>>>(x, mk, mv, mr, xk, xv, xr);

  dim3 ggrid(D_EMB / 128, M_TOT / 128);   // (8, 256)
  gemm_bf16_wmma<<<ggrid, 256, 0, stream>>>(xk, wkb, Kf, M_TOT, D_EMB, D_EMB);
  gemm_bf16_wmma<<<ggrid, 256, 0, stream>>>(xv, wvb, Vf, M_TOT, D_EMB, D_EMB);
  gemm_bf16_wmma<<<ggrid, 256, 0, stream>>>(xr, wrb, Rf, M_TOT, D_EMB, D_EMB);

  const unsigned PB = (B_SZ * NCHUNK * D_EMB) / 256;  // 1024 blocks
  wkv_pass1<<<PB, 256, 0, stream>>>(Kf, Vf, ld, Sc, Sn);
  wkv_combine<<<(B_SZ * D_EMB) / 256, 256, 0, stream>>>(Sc, Sn, ld, Ic, In);
  wkv_pass2<<<PB, 256, 0, stream>>>(Kf, Vf, Rf, ld, lg, Ic, In, rwkv);

  gemm_bf16_wmma<<<ggrid, 256, 0, stream>>>(rwkv, wob, out, M_TOT, D_EMB, D_EMB);
}
